// DiffPool_498216206442
// MI455X (gfx1250) — matrix operand order repaired and checked by reading, not compiled, not run on hardware
//
#include <hip/hip_runtime.h>
#include <hip/hip_bf16.h>
#include <math.h>

// DiffPool forward for MI455X (gfx1250, wave32, WMMA bf16->f32).
// N=8192, F_IN=N_EMBED=N_META=128.
#define NV 8192
#define FD 128

typedef __attribute__((ext_vector_type(16))) __bf16 bf16x16;
typedef __attribute__((ext_vector_type(8)))  float  fx8;
typedef __attribute__((ext_vector_type(4)))  float  fx4;   // clang vector: valid for nontemporal builtins

union BfVec { bf16x16 v; uint4 u[2]; };

__device__ __forceinline__ fx8 wmma_bf16(bf16x16 a, bf16x16 b, fx8 c) {
  // D = A(16x32) * B(32x16) + C, f32 accumulate
  return __builtin_amdgcn_wmma_f32_16x16x32_bf16(false, a, false, b, (short)0, c, false, false);
}

// Pin per-iteration schedule: [VMEM loads] -> [cvt VALU] -> [back-to-back WMMAs].
__device__ __forceinline__ void sched_loads_then_wmma() {
#if __has_builtin(__builtin_amdgcn_sched_group_barrier)
  __builtin_amdgcn_sched_group_barrier(0x020, 20, 0);  // VMEM reads (16 B-frag + 4 A)
  __builtin_amdgcn_sched_group_barrier(0x002, 16, 0);  // VALU (v_cvt_pk_bf16_f32 etc.)
  __builtin_amdgcn_sched_group_barrier(0x008, 8, 0);   // 8 consecutive WMMAs
#endif
}

// ---- fragment loaders (ISA 7.12.2 layouts, wave32) ----
// A-frag (16x32, bf16): row = rowBase + (lane&15); K(e) = 8*(lane>>4) + (e<8 ? e : e+8)
// B-frag (32x16, bf16): col = colBase + (lane&15); K(e) = 8*(lane>>4) + (e<8 ? e : e+8)
// C/D   (16x16, f32) :  M = r + 8*(lane>>4); N = lane&15

__device__ __forceinline__ bf16x16 cvt16(fx4 c0, fx4 c1, fx4 c2, fx4 c3) {
  bf16x16 a;
  a[0]=(__bf16)c0[0];  a[1]=(__bf16)c0[1];  a[2]=(__bf16)c0[2];  a[3]=(__bf16)c0[3];
  a[4]=(__bf16)c1[0];  a[5]=(__bf16)c1[1];  a[6]=(__bf16)c1[2];  a[7]=(__bf16)c1[3];
  a[8]=(__bf16)c2[0];  a[9]=(__bf16)c2[1];  a[10]=(__bf16)c2[2]; a[11]=(__bf16)c2[3];
  a[12]=(__bf16)c3[0]; a[13]=(__bf16)c3[1]; a[14]=(__bf16)c3[2]; a[15]=(__bf16)c3[3];
  return a;
}

// streamed f32 source (A / AX): non-temporal — single-touch data, keep it out of L2
__device__ __forceinline__ bf16x16 load_a_f32_nt(const float* __restrict__ src, size_t ld,
                                                 int rowBase, int kk, int l15, int lh) {
  const float* p = src + (size_t)(rowBase + l15) * ld + kk + 8 * lh;   // 32B aligned
  fx4 c0 = __builtin_nontemporal_load((const fx4*)p);
  fx4 c1 = __builtin_nontemporal_load(((const fx4*)p) + 1);
  fx4 c2 = __builtin_nontemporal_load((const fx4*)(p + 16));
  fx4 c3 = __builtin_nontemporal_load(((const fx4*)(p + 16)) + 1);
  return cvt16(c0, c1, c2, c3);
}

__device__ __forceinline__ bf16x16 load_a_bf16(const __bf16* __restrict__ src, size_t ld,
                                               int rowBase, int kk, int l15, int lh) {
  const __bf16* p = src + (size_t)(rowBase + l15) * ld + kk + 8 * lh;  // 16B aligned chunks
  BfVec r;
  r.u[0] = *(const uint4*)p;
  r.u[1] = *(const uint4*)(p + 16);
  return r.v;
}

__device__ __forceinline__ void load_b_into(BfVec& dst, const __bf16* __restrict__ srcT,
                                            size_t ld, int colBase, int kk, int l15, int lh) {
  const __bf16* p = srcT + (size_t)(colBase + l15) * ld + kk + 8 * lh; // 16B aligned chunks
  dst.u[0] = *(const uint4*)p;
  dst.u[1] = *(const uint4*)(p + 16);
}

// ---------------- pass 1: degrees -> d^-1/2 ----------------
__global__ void k_deg(const float* __restrict__ A, float* __restrict__ dinv) {
  __shared__ float red[256];
  int row = blockIdx.x;
  const float4* p = (const float4*)(A + (size_t)row * NV);
  float s = 0.f;
  for (int k = threadIdx.x; k < NV / 4; k += 256) { float4 v = p[k]; s += v.x + v.y + v.z + v.w; }
  red[threadIdx.x] = s;
  __syncthreads();
  for (int off = 128; off > 0; off >>= 1) {
    if ((int)threadIdx.x < off) red[threadIdx.x] += red[threadIdx.x + off];
    __syncthreads();
  }
  if (threadIdx.x == 0) {
    float deg = red[0] + 1.0f;                   // A_hat = A + I
    dinv[row] = deg > 0.f ? rsqrtf(deg) : 0.f;
  }
}

// ---------------- pass 2: XsT[n][j] = bf16(dinv[j]*X[j][n]) ----------------
__global__ void k_xst(const float* __restrict__ X, const float* __restrict__ dinv,
                      __bf16* __restrict__ XsT) {
  int idx = blockIdx.x * 256 + threadIdx.x;      // 0 .. N*FD-1, idx = n*NV + j
  int j = idx & (NV - 1);
  int n = idx >> 13;
  XsT[idx] = (__bf16)(dinv[j] * X[(size_t)j * FD + n]);
}

__global__ void k_wcvt(const float* __restrict__ We, const float* __restrict__ Wa,
                       __bf16* __restrict__ WbE, __bf16* __restrict__ WbA) {
  int idx = blockIdx.x * 256 + threadIdx.x;      // 128*128
  WbE[idx] = (__bf16)We[idx];
  WbA[idx] = (__bf16)Wa[idx];
}

// ---------------- pass 3: AX = dinv .* (A @ Xs + Xs) ----------------
__global__ __launch_bounds__(256) void k_gemm_ax(const float* __restrict__ A,
                                                 const __bf16* __restrict__ XsT,
                                                 const float* __restrict__ dinv,
                                                 float* __restrict__ AX) {
  int wave = threadIdx.x >> 5, lane = threadIdx.x & 31;
  int l15 = lane & 15, lh = lane >> 4;
  int strip = blockIdx.x * 128 + wave * 16;      // 16 output rows per wave
  fx8 acc[8] = {};
  for (int kk = 0; kk < NV; kk += 32) {
    // preload all 8 B fragments (one clause, one wait) ...
    BfVec b[8];
#pragma unroll
    for (int f = 0; f < 8; ++f) load_b_into(b[f], XsT, NV, 16 * f, kk, l15, lh);
    const float* ap = A + (size_t)(strip + l15) * NV + kk + 8 * lh;
    if (kk + 544 <= NV) __builtin_prefetch(ap + 512, 0, 3);   // global_prefetch_b8
    bf16x16 aF = load_a_f32_nt(A, NV, strip, kk, l15, lh);
    // ... then 8 back-to-back WMMAs
#pragma unroll
    for (int f = 0; f < 8; ++f) acc[f] = wmma_bf16(aF, b[f].v, acc[f]);
    sched_loads_then_wmma();
  }
#pragma unroll
  for (int r = 0; r < 8; ++r) {
    int i = strip + r + 8 * lh;
    float di = dinv[i];
#pragma unroll
    for (int f = 0; f < 8; ++f) {
      int n = 16 * f + l15;
      float xs = (float)XsT[(size_t)n * NV + i];      // identity term dinv[i]^2 * X[i,n]
      AX[(size_t)i * FD + n] = di * (acc[f][r] + xs);
    }
  }
}

// ---------------- pass 4: z = AX@We^T+be ; s = softmax(AX@Wa^T+ba) ----------------
__global__ __launch_bounds__(256) void k_gemm_zs(const float* __restrict__ AX,
                                                 const __bf16* __restrict__ WbE,
                                                 const __bf16* __restrict__ WbA,
                                                 const float* __restrict__ be,
                                                 const float* __restrict__ ba,
                                                 float* __restrict__ out_s,
                                                 __bf16* __restrict__ zT,
                                                 __bf16* __restrict__ SlT) {
  __shared__ __bf16 lT[128 * 136];               // padded: stride 136 elems = 272B, conflict-free
  int wave = threadIdx.x >> 5, lane = threadIdx.x & 31;
  int l15 = lane & 15, lh = lane >> 4;
  int strip0 = blockIdx.x * 128;
  int strip = strip0 + wave * 16;
  fx8 accZ[8] = {}, accS[8] = {};
#pragma unroll
  for (int kk = 0; kk < FD; kk += 32) {
    bf16x16 aF = load_a_f32_nt(AX, FD, strip, kk, l15, lh);
    BfVec b[8];
#pragma unroll
    for (int f = 0; f < 8; ++f) load_b_into(b[f], WbE, FD, 16 * f, kk, l15, lh);
#pragma unroll
    for (int f = 0; f < 8; ++f) accZ[f] = wmma_bf16(aF, b[f].v, accZ[f]);
#pragma unroll
    for (int f = 0; f < 8; ++f) load_b_into(b[f], WbA, FD, 16 * f, kk, l15, lh);
#pragma unroll
    for (int f = 0; f < 8; ++f) accS[f] = wmma_bf16(aF, b[f].v, accS[f]);
  }
  // biases (per output column n = 16f + l15, same for every row r)
#pragma unroll
  for (int f = 0; f < 8; ++f) {
    float bz = be[16 * f + l15], bs = ba[16 * f + l15];
#pragma unroll
    for (int r = 0; r < 8; ++r) { accZ[f][r] += bz; accS[f][r] += bs; }
  }
  // softmax over each row: shfl_xor masks <=8 stay within the 16-lane half holding one row
#pragma unroll
  for (int r = 0; r < 8; ++r) {
    float m = -1e30f;
#pragma unroll
    for (int f = 0; f < 8; ++f) m = fmaxf(m, accS[f][r]);
#pragma unroll
    for (int off = 8; off >= 1; off >>= 1) m = fmaxf(m, __shfl_xor(m, off, 32));
    float s = 0.f;
#pragma unroll
    for (int f = 0; f < 8; ++f) { float e = __expf(accS[f][r] - m); accS[f][r] = e; s += e; }
#pragma unroll
    for (int off = 8; off >= 1; off >>= 1) s += __shfl_xor(s, off, 32);
    float inv = 1.0f / s;
#pragma unroll
    for (int f = 0; f < 8; ++f) accS[f][r] *= inv;
  }
  // s_l (f32) -> d_out
#pragma unroll
  for (int r = 0; r < 8; ++r) {
    int i = strip + r + 8 * lh;
#pragma unroll
    for (int f = 0; f < 8; ++f)
      out_s[(size_t)i * FD + 16 * f + l15] = accS[f][r];
  }
  // transpose s -> SlT (bf16) via LDS, coalesced global writes
#pragma unroll
  for (int f = 0; f < 8; ++f)
#pragma unroll
    for (int r = 0; r < 8; ++r)
      lT[(16 * f + l15) * 136 + wave * 16 + r + 8 * lh] = (__bf16)accS[f][r];
  __syncthreads();
  {
    int col = threadIdx.x >> 1, half = threadIdx.x & 1;
    const uint4* src = (const uint4*)(&lT[col * 136 + half * 64]);
    uint4* dst = (uint4*)(SlT + (size_t)col * NV + strip0 + half * 64);
#pragma unroll
    for (int q = 0; q < 8; ++q) dst[q] = src[q];
  }
  __syncthreads();
  // transpose z -> zT (bf16)
#pragma unroll
  for (int f = 0; f < 8; ++f)
#pragma unroll
    for (int r = 0; r < 8; ++r)
      lT[(16 * f + l15) * 136 + wave * 16 + r + 8 * lh] = (__bf16)accZ[f][r];
  __syncthreads();
  {
    int col = threadIdx.x >> 1, half = threadIdx.x & 1;
    const uint4* src = (const uint4*)(&lT[col * 136 + half * 64]);
    uint4* dst = (uint4*)(zT + (size_t)col * NV + strip0 + half * 64);
#pragma unroll
    for (int q = 0; q < 8; ++q) dst[q] = src[q];
  }
}

// ---------------- pass 5: B2T = (A @ s_l)^T (bf16) ----------------
__global__ __launch_bounds__(256) void k_gemm_as(const float* __restrict__ A,
                                                 const __bf16* __restrict__ SlT,
                                                 __bf16* __restrict__ B2T) {
  __shared__ __bf16 lT[128 * 136];
  int wave = threadIdx.x >> 5, lane = threadIdx.x & 31;
  int l15 = lane & 15, lh = lane >> 4;
  int strip0 = blockIdx.x * 128;
  int strip = strip0 + wave * 16;
  fx8 acc[8] = {};
  for (int kk = 0; kk < NV; kk += 32) {
    BfVec b[8];
#pragma unroll
    for (int f = 0; f < 8; ++f) load_b_into(b[f], SlT, NV, 16 * f, kk, l15, lh);
    const float* ap = A + (size_t)(strip + l15) * NV + kk + 8 * lh;
    if (kk + 544 <= NV) __builtin_prefetch(ap + 512, 0, 3);
    bf16x16 aF = load_a_f32_nt(A, NV, strip, kk, l15, lh);
#pragma unroll
    for (int f = 0; f < 8; ++f) acc[f] = wmma_bf16(aF, b[f].v, acc[f]);
    sched_loads_then_wmma();
  }
#pragma unroll
  for (int f = 0; f < 8; ++f)
#pragma unroll
    for (int r = 0; r < 8; ++r)
      lT[(16 * f + l15) * 136 + wave * 16 + r + 8 * lh] = (__bf16)acc[f][r];
  __syncthreads();
  {
    int col = threadIdx.x >> 1, half = threadIdx.x & 1;
    const uint4* src = (const uint4*)(&lT[col * 136 + half * 64]);
    uint4* dst = (uint4*)(B2T + (size_t)col * NV + strip0 + half * 64);
#pragma unroll
    for (int q = 0; q < 8; ++q) dst[q] = src[q];
  }
}

__global__ void k_zero(float* __restrict__ out) {
  out[blockIdx.x * 256 + threadIdx.x] = 0.f;     // X_next + A_next region (32768 floats)
}

// ---------------- pass 6: X_next = Sl^T @ z ; A_next = Sl^T @ B2 (split-K, atomics) --------
__global__ __launch_bounds__(256) void k_final(const __bf16* __restrict__ SlT,
                                               const __bf16* __restrict__ zT,
                                               const __bf16* __restrict__ B2T,
                                               float* __restrict__ out) {
  int wave = threadIdx.x >> 5, lane = threadIdx.x & 31;
  int l15 = lane & 15, lh = lane >> 4;
  int mBase = wave * 16;                         // 8 waves cover M = 128
  int k0 = blockIdx.x * 128;                     // 64 blocks split K = 8192
  fx8 accX[8] = {}, accA[8] = {};
#pragma unroll
  for (int kk = 0; kk < 128; kk += 32) {
    bf16x16 aF = load_a_bf16(SlT, NV, mBase, k0 + kk, l15, lh);
    BfVec b[8];
#pragma unroll
    for (int f = 0; f < 8; ++f) load_b_into(b[f], zT, NV, 16 * f, k0 + kk, l15, lh);
#pragma unroll
    for (int f = 0; f < 8; ++f) accX[f] = wmma_bf16(aF, b[f].v, accX[f]);
#pragma unroll
    for (int f = 0; f < 8; ++f) load_b_into(b[f], B2T, NV, 16 * f, k0 + kk, l15, lh);
#pragma unroll
    for (int f = 0; f < 8; ++f) accA[f] = wmma_bf16(aF, b[f].v, accA[f]);
  }
#pragma unroll
  for (int r = 0; r < 8; ++r) {
    int m = mBase + r + 8 * lh;
#pragma unroll
    for (int f = 0; f < 8; ++f) {
      int n = 16 * f + l15;
      atomicAdd(&out[m * FD + n], accX[f][r]);           // X_next
      atomicAdd(&out[16384 + m * FD + n], accA[f][r]);   // A_next
    }
  }
}

extern "C" void kernel_launch(void* const* d_in, const int* in_sizes, int n_in,
                              void* d_out, int out_size, void* d_ws, size_t ws_size,
                              hipStream_t stream) {
  (void)in_sizes; (void)n_in; (void)out_size; (void)ws_size;
  const float* X  = (const float*)d_in[0];
  const float* A  = (const float*)d_in[1];
  const float* We = (const float*)d_in[2];
  const float* be = (const float*)d_in[3];
  const float* Wa = (const float*)d_in[4];
  const float* ba = (const float*)d_in[5];
  float* out = (float*)d_out;

  // workspace layout (~12.1 MB)
  char* ws = (char*)d_ws;
  float*  dinv = (float*) (ws + 0);                      //  32 KB
  __bf16* XsT  = (__bf16*)(ws + 32768);                  //   2 MB  [128][8192]
  __bf16* WbE  = (__bf16*)(ws + 2129920);                //  32 KB
  __bf16* WbA  = (__bf16*)(ws + 2162688);                //  32 KB
  float*  AX   = (float*) (ws + 2195456);                //   4 MB  [8192][128]
  __bf16* zT   = (__bf16*)(ws + 6389760);                //   2 MB  [128][8192]
  __bf16* SlT  = (__bf16*)(ws + 8486912);                //   2 MB  [128][8192]
  __bf16* B2T  = (__bf16*)(ws + 10584064);               //   2 MB  [128][8192]

  k_deg    <<<8192, 256, 0, stream>>>(A, dinv);
  k_xst    <<<4096, 256, 0, stream>>>(X, dinv, XsT);
  k_wcvt   <<<64,   256, 0, stream>>>(We, Wa, WbE, WbA);
  k_gemm_ax<<<64,   256, 0, stream>>>(A, XsT, dinv, AX);
  k_gemm_zs<<<64,   256, 0, stream>>>(AX, WbE, WbA, be, ba, out + 32768, zT, SlT);
  k_gemm_as<<<64,   256, 0, stream>>>(A, SlT, B2T);
  k_zero   <<<128,  256, 0, stream>>>(out);
  k_final  <<<64,   256, 0, stream>>>(SlT, zT, B2T, out);
}
